// SDGraphEncoder_1821066133673
// MI455X (gfx1250) — matrix operand order repaired and checked by reading, not compiled, not run on hardware
//
#include <hip/hip_runtime.h>
#include <cstdint>
#include <cstddef>

typedef __bf16 bf16;
typedef __attribute__((ext_vector_type(16))) __bf16 v16bf;
typedef __attribute__((ext_vector_type(8)))  __bf16 v8bf;
typedef __attribute__((ext_vector_type(8)))  float  v8f;

static __device__ __forceinline__ float gelu_f(float x) {
  const float c = 0.7978845608028654f;
  float x3 = x * x * x;
  return 0.5f * x * (1.0f + tanhf(c * (x + 0.044715f * x3)));
}

// ---------------------------------------------------------------------------
// Generic bf16 WMMA GEMM:  C[M][N] = A[M][K] * B  with B stored as X[N][K]
// (K contiguous per output column). One wave -> 16x64 C tile, 4 accumulators.
// A-frag: lane m=lane&15, half-select
// hs=lane>>4 picks 16 contiguous K at hs*16 (same k order as B-frag).
// B-frag (same k order as A): lane n=lane&15, 16 contiguous K at hs*16.
// ---------------------------------------------------------------------------
__global__ void __launch_bounds__(256) gemm_bf16(
    const bf16* __restrict__ A, const bf16* __restrict__ B,
    void* __restrict__ C, int M, int N, int K, int ldc, int c_is_bf16)
{
  int wid  = blockIdx.x * 8 + (threadIdx.x >> 5);
  int lane = threadIdx.x & 31;
  int mt = M >> 4, nt = N >> 6;
  if (wid >= mt * nt) return;
  int tm = wid % mt, tn = wid / mt;
  int lrow = lane & 15;
  int hs   = lane >> 4;

  const bf16* arow = A + (size_t)(tm * 16 + lrow) * (size_t)K;
  v8f acc0 = {}, acc1 = {}, acc2 = {}, acc3 = {};

  for (int k0 = 0; k0 < K; k0 += 32) {
    v8bf alo = *(const v8bf*)(arow + k0 + hs * 16);
    v8bf ahi = *(const v8bf*)(arow + k0 + hs * 16 + 8);
    v16bf a;
#pragma unroll
    for (int i = 0; i < 8; i++) { a[i] = alo[i]; a[8 + i] = ahi[i]; }

    const bf16* bbase = B + (size_t)(tn * 64 + lrow) * (size_t)K + k0 + hs * 16;
    v16bf b0 = *(const v16bf*)(bbase + (size_t)0  * K);
    v16bf b1 = *(const v16bf*)(bbase + (size_t)16 * K);
    v16bf b2 = *(const v16bf*)(bbase + (size_t)32 * K);
    v16bf b3 = *(const v16bf*)(bbase + (size_t)48 * K);

    acc0 = __builtin_amdgcn_wmma_f32_16x16x32_bf16(false, a, false, b0, (short)0, acc0, false, false);
    acc1 = __builtin_amdgcn_wmma_f32_16x16x32_bf16(false, a, false, b1, (short)0, acc1, false, false);
    acc2 = __builtin_amdgcn_wmma_f32_16x16x32_bf16(false, a, false, b2, (short)0, acc2, false, false);
    acc3 = __builtin_amdgcn_wmma_f32_16x16x32_bf16(false, a, false, b3, (short)0, acc3, false, false);
  }

  int n0 = tn * 64 + lrow;
#pragma unroll
  for (int r = 0; r < 8; r++) {
    int m = tm * 16 + r + hs * 8;
    size_t base = (size_t)m * (size_t)ldc;
    if (c_is_bf16) {
      bf16* c = (bf16*)C;
      c[base + n0]      = (bf16)acc0[r];
      c[base + n0 + 16] = (bf16)acc1[r];
      c[base + n0 + 32] = (bf16)acc2[r];
      c[base + n0 + 48] = (bf16)acc3[r];
    } else {
      float* c = (float*)C;
      c[base + n0]      = acc0[r];
      c[base + n0 + 16] = acc1[r];
      c[base + n0 + 32] = acc2[r];
      c[base + n0 + 48] = acc3[r];
    }
  }
}

// ---------------------------------------------------------------------------
// BN batch statistics: one block per channel; stat[ch]=mean, stat[512+ch]=rstd
// ---------------------------------------------------------------------------
__global__ void __launch_bounds__(256) bn_stats_f32(
    const float* __restrict__ Y, int N, float* __restrict__ stat)
{
  int ch = blockIdx.x, t = threadIdx.x;
  const float* row = Y + (size_t)ch * (size_t)N;
  float s = 0.f, s2 = 0.f;
  for (int i = t; i < N; i += 256) { float v = row[i]; s += v; s2 += v * v; }
  __shared__ float sh[256], sh2[256];
  sh[t] = s; sh2[t] = s2; __syncthreads();
  for (int o = 128; o > 0; o >>= 1) {
    if (t < o) { sh[t] += sh[t + o]; sh2[t] += sh2[t + o]; }
    __syncthreads();
  }
  if (t == 0) {
    float m = sh[0] / (float)N;
    float var = sh2[0] / (float)N - m * m;
    stat[ch] = m;
    stat[512 + ch] = rsqrtf(var + 1e-5f);
  }
}

__global__ void __launch_bounds__(256) bn_stats_bf16(
    const bf16* __restrict__ Y, int N, float* __restrict__ stat)
{
  int ch = blockIdx.x, t = threadIdx.x;
  const bf16* row = Y + (size_t)ch * (size_t)N;
  float s = 0.f, s2 = 0.f;
  for (int i = t; i < N; i += 256) { float v = (float)row[i]; s += v; s2 += v * v; }
  __shared__ float sh[256], sh2[256];
  sh[t] = s; sh2[t] = s2; __syncthreads();
  for (int o = 128; o > 0; o >>= 1) {
    if (t < o) { sh[t] += sh[t + o]; sh2[t] += sh2[t + o]; }
    __syncthreads();
  }
  if (t == 0) {
    float m = sh[0] / (float)N;
    float var = sh2[0] / (float)N - m * m;
    stat[ch] = m;
    stat[512 + ch] = rsqrtf(var + 1e-5f);
  }
}

// ---------------------------------------------------------------------------
// Elementwise / packing / graph kernels
// ---------------------------------------------------------------------------
__global__ void __launch_bounds__(256) cvt_bf16(
    const float* __restrict__ src, bf16* __restrict__ dst, int n)
{
  int i = blockIdx.x * 256 + threadIdx.x;
  if (i < n) dst[i] = (bf16)src[i];
}

// im2col for temporal (1,3) conv: X[(b*64+s)*32+p][i*3+kk], pad over p
__global__ void __launch_bounds__(256) im2col_t(
    const float* __restrict__ df, bf16* __restrict__ X)
{
  long long idx = (long long)blockIdx.x * 256 + threadIdx.x;
  if (idx >= 16384LL * 384) return;
  int k = (int)(idx % 384); int col = (int)(idx / 384);
  int p = col & 31, s = (col >> 5) & 63, b = col >> 11;
  int kk = k % 3, i = k / 3;
  int pp = p + kk - 1;
  float v = (pp >= 0 && pp < 32) ? df[(((size_t)b * 128 + i) * 64 + s) * 32 + pp] : 0.f;
  X[idx] = (bf16)v;
}

// temporal post: BN+GELU, max over p -> Xc[:,256:384] (bf16)
__global__ void __launch_bounds__(256) apply_t_max(
    const float* __restrict__ Y, const float* __restrict__ stat,
    const float* __restrict__ g, const float* __restrict__ bt, bf16* __restrict__ Xc)
{
  int idx = blockIdx.x * 256 + threadIdx.x;
  if (idx >= 512 * 128) return;
  int ch = idx & 127, cbs = idx >> 7;
  float m = stat[ch], r = stat[512 + ch], gg = g[ch], bb = bt[ch];
  const float* row = Y + (size_t)ch * 16384 + (size_t)cbs * 32;
  float mx = -3.0e38f;
  for (int p = 0; p < 32; p++) {
    float v = gelu_f((row[p] - m) * r * gg + bb);
    mx = fmaxf(mx, v);
  }
  Xc[(size_t)cbs * 384 + 256 + ch] = (bf16)mx;
}

__global__ void __launch_bounds__(256) pack_sparse_xc(
    const float* __restrict__ sf, bf16* __restrict__ Xc)
{
  int idx = blockIdx.x * 256 + threadIdx.x;
  if (idx >= 512 * 256) return;
  int c = idx & 255, cbs = idx >> 8;
  int b = cbs >> 6, s = cbs & 63;
  Xc[(size_t)cbs * 384 + c] = (bf16)sf[((size_t)b * 256 + c) * 64 + s];
}

// BN+GELU with [ch][col] -> [col][ch] transpose, f32 out
__global__ void __launch_bounds__(256) apply_bn_gelu_T_f32(
    const float* __restrict__ Y, int M, int N, const float* __restrict__ stat,
    const float* __restrict__ g, const float* __restrict__ bt, float* __restrict__ out)
{
  long long idx = (long long)blockIdx.x * 256 + threadIdx.x;
  if (idx >= (long long)M * N) return;
  int c = (int)(idx % M);
  long long col = idx / M;
  float v = Y[(size_t)c * (size_t)N + col];
  v = gelu_f((v - stat[c]) * stat[512 + c] * g[c] + bt[c]);
  out[(size_t)col * M + c] = v;
}

// same, bf16 out
__global__ void __launch_bounds__(256) apply_bn_gelu_T_bf16(
    const float* __restrict__ Y, int M, int N, const float* __restrict__ stat,
    const float* __restrict__ g, const float* __restrict__ bt, bf16* __restrict__ out)
{
  long long idx = (long long)blockIdx.x * 256 + threadIdx.x;
  if (idx >= (long long)M * N) return;
  int c = (int)(idx % M);
  long long col = idx / M;
  float v = Y[(size_t)c * (size_t)N + col];
  v = gelu_f((v - stat[c]) * stat[512 + c] * g[c] + bt[c]);
  out[(size_t)col * M + c] = (bf16)v;
}

// concat [dense(128), broadcast sparse(256)] -> Xd[(b,s,p)][384]
__global__ void __launch_bounds__(256) build_xd(
    const float* __restrict__ df, const float* __restrict__ sf, bf16* __restrict__ Xd)
{
  long long idx = (long long)blockIdx.x * 256 + threadIdx.x;
  if (idx >= 16384LL * 384) return;
  int k = (int)(idx % 384); int col = (int)(idx / 384);
  int p = col & 31, s = (col >> 5) & 63, b = col >> 11;
  float v = (k < 128) ? df[(((size_t)b * 128 + k) * 64 + s) * 32 + p]
                      : sf[((size_t)b * 256 + (k - 128)) * 64 + s];
  Xd[idx] = (bf16)v;
}

__global__ void __launch_bounds__(256) sqnorm_k(
    const bf16* __restrict__ udn, float* __restrict__ nn)
{
  int t = blockIdx.x * 256 + threadIdx.x;
  if (t >= 16384) return;
  const bf16* row = udn + (size_t)t * 128;
  float s = 0.f;
  for (int c = 0; c < 128; c++) { float v = (float)row[c]; s += v * v; }
  nn[t] = s;
}

// per-batch top-10 nearest (smallest d), G = gram for this batch
__global__ void __launch_bounds__(256) topk10(
    const float* __restrict__ G, const float* __restrict__ nnb, int* __restrict__ out)
{
  int n = blockIdx.x * 256 + threadIdx.x;
  if (n >= 2048) return;
  float best[10]; int bi[10];
#pragma unroll
  for (int j = 0; j < 10; j++) { best[j] = 3.0e38f; bi[j] = 0; }
  float dn_ = nnb[n];
  const float* grow = G + (size_t)n * 2048;
  for (int mm = 0; mm < 2048; mm++) {
    float d = dn_ + nnb[mm] - 2.f * grow[mm];
    if (d < best[9]) {
      int j = 9;
      while (j > 0 && best[j - 1] > d) { best[j] = best[j - 1]; bi[j] = bi[j - 1]; j--; }
      best[j] = d; bi[j] = mm;
    }
  }
  for (int j = 0; j < 10; j++) out[(size_t)n * 10 + j] = bi[j];
}

// per-batch dense edge features: Xe[(n*10+kk)][256]
__global__ void __launch_bounds__(256) build_edge_dn(
    const bf16* __restrict__ udn_b, const int* __restrict__ idx_b, bf16* __restrict__ Xe)
{
  long long idx = (long long)blockIdx.x * 256 + threadIdx.x;
  if (idx >= 20480LL * 256) return;
  int ch = (int)(idx & 255); int col = (int)(idx >> 8);
  int kk = col % 10, n = col / 10;
  int nb = idx_b[(size_t)n * 10 + kk];
  int c = ch & 127;
  float ctr = (float)udn_b[(size_t)n * 128 + c];
  float v = (ch < 128) ? ((float)udn_b[(size_t)nb * 128 + c] - ctr) : ctr;
  Xe[idx] = (bf16)v;
}

// dense edge post: BN+GELU over bf16 Y [256][163840], max over kk=10 -> dnf f32
__global__ void __launch_bounds__(256) apply_dn_max(
    const bf16* __restrict__ Y, const float* __restrict__ stat,
    const float* __restrict__ g, const float* __restrict__ bt, float* __restrict__ dnf)
{
  long long idx = (long long)blockIdx.x * 256 + threadIdx.x;
  if (idx >= 8LL * 2048 * 256) return;
  int c = (int)(idx & 255);
  long long ng = idx >> 8;
  float m = stat[c], r = stat[512 + c], gg = g[c], bb = bt[c];
  const bf16* yb = Y + (size_t)c * 163840 + (size_t)ng * 10;
  float mx = -3.0e38f;
  for (int kk = 0; kk < 10; kk++) {
    float v = gelu_f(((float)yb[kk] - m) * r * gg + bb);
    mx = fmaxf(mx, v);
  }
  dnf[(size_t)ng * 256 + c] = mx;
}

// sparse feature-space knn (k=2 incl. self)
__global__ void __launch_bounds__(256) knn_sp(
    const float* __restrict__ usp, int* __restrict__ idxs)
{
  int t = blockIdx.x * 256 + threadIdx.x;
  if (t >= 512) return;
  int b = t >> 6, n = t & 63;
  const float* xb = usp + (size_t)b * 64 * 256;
  const float* xn = xb + (size_t)n * 256;
  float b0 = 3.0e38f, b1 = 3.0e38f; int i0 = 0, i1 = 0;
  for (int mm = 0; mm < 64; mm++) {
    const float* xm = xb + (size_t)mm * 256;
    float s = 0.f;
    for (int c = 0; c < 256; c++) { float d = xn[c] - xm[c]; s += d * d; }
    if (s < b0) { b1 = b0; i1 = i0; b0 = s; i0 = mm; }
    else if (s < b1) { b1 = s; i1 = mm; }
  }
  idxs[t * 2] = i0; idxs[t * 2 + 1] = i1;
}

__global__ void __launch_bounds__(256) build_edge_sp_k(
    const float* __restrict__ usp, const int* __restrict__ idxs, bf16* __restrict__ X)
{
  long long idx = (long long)blockIdx.x * 256 + threadIdx.x;
  if (idx >= 1024LL * 512) return;
  int ch = (int)(idx & 511); int col = (int)(idx >> 9);
  int kk = col & 1, t = col >> 1;
  int b = t >> 6, n = t & 63;
  int nb = idxs[t * 2 + kk];
  int c = ch & 255;
  const float* xb = usp + (size_t)b * 64 * 256;
  float ctr = xb[(size_t)n * 256 + c];
  float v = (ch < 256) ? (xb[(size_t)nb * 256 + c] - ctr) : ctr;
  X[(size_t)col * 512 + ch] = (bf16)v;
}

// sparse edge post: Y [512][1024], max over kk=2 -> usp2 [b][n][512]
__global__ void __launch_bounds__(256) apply_sp_max(
    const float* __restrict__ Y, const float* __restrict__ stat,
    const float* __restrict__ g, const float* __restrict__ bt, float* __restrict__ usp2)
{
  int idx = blockIdx.x * 256 + threadIdx.x;
  if (idx >= 8 * 64 * 512) return;
  int c = idx & 511, t = idx >> 9;
  float m = stat[c], r = stat[512 + c], gg = g[c], bb = bt[c];
  float v0 = gelu_f((Y[(size_t)c * 1024 + t * 2 + 0] - m) * r * gg + bb);
  float v1 = gelu_f((Y[(size_t)c * 1024 + t * 2 + 1] - m) * r * gg + bb);
  usp2[(size_t)t * 512 + c] = fmaxf(v0, v1);
}

// farthest point sampling, one block per batch, 64 threads
__global__ void fps_k(const float* __restrict__ coor, int* __restrict__ fidx)
{
  int b = blockIdx.x, n = threadIdx.x;
  __shared__ float sd[64];
  __shared__ int slast;
  const float* cb = coor + (size_t)b * 64 * 32;
  float d = 1e10f;
  int last = 0;
  for (int m = 0; m < 32; m++) {
    if (n == 0) fidx[b * 32 + m] = last;
    float s = 0.f;
    for (int c = 0; c < 32; c++) {
      float df = cb[(size_t)n * 32 + c] - cb[(size_t)last * 32 + c];
      s += df * df;
    }
    d = fminf(d, s);
    sd[n] = d;
    __syncthreads();
    if (n == 0) {
      float bv = -1.f; int bi = 0;
      for (int i = 0; i < 64; i++) if (sd[i] > bv) { bv = sd[i]; bi = i; }
      slast = bi;
    }
    __syncthreads();
    last = slast;
  }
}

// knn (k=2) on stroke coords, only at sampled centers
__global__ void __launch_bounds__(256) knn_coor_k(
    const float* __restrict__ coor, const int* __restrict__ fidx, int* __restrict__ knnf)
{
  int t = blockIdx.x * 256 + threadIdx.x;
  if (t >= 256) return;
  int b = t >> 5, m = t & 31;
  int ctr = fidx[b * 32 + m];
  const float* cb = coor + (size_t)b * 2048;
  float b0 = 3.0e38f, b1 = 3.0e38f; int i0 = 0, i1 = 0;
  for (int j = 0; j < 64; j++) {
    float s = 0.f;
    for (int c = 0; c < 32; c++) {
      float df = cb[(size_t)ctr * 32 + c] - cb[(size_t)j * 32 + c];
      s += df * df;
    }
    if (s < b0) { b1 = b0; i1 = i0; b0 = s; i0 = j; }
    else if (s < b1) { b1 = s; i1 = j; }
  }
  knnf[t * 2] = i0; knnf[t * 2 + 1] = i1;
}

__global__ void __launch_bounds__(256) gather_coor_k(
    const float* __restrict__ coor, const int* __restrict__ fidx, float* __restrict__ out)
{
  int idx = blockIdx.x * 256 + threadIdx.x;
  if (idx >= 8192) return;
  int d = idx & 31, m = (idx >> 5) & 31, b = idx >> 10;
  out[idx] = coor[(size_t)b * 2048 + (size_t)fidx[b * 32 + m] * 32 + d];
}

__global__ void __launch_bounds__(256) build_edge_ds_sp(
    const float* __restrict__ usp2, const int* __restrict__ fidx,
    const int* __restrict__ knnf, bf16* __restrict__ X)
{
  long long idx = (long long)blockIdx.x * 256 + threadIdx.x;
  if (idx >= 512LL * 1024) return;
  int ch = (int)(idx & 1023); int col = (int)(idx >> 10);
  int kk = col & 1, t = col >> 1;
  int b = t >> 5;
  int nbr = knnf[t * 2 + kk], ctr = fidx[t];
  int c = ch & 511;
  const float* ub = usp2 + (size_t)b * 64 * 512;
  float cv = ub[(size_t)ctr * 512 + c];
  float v = (ch < 512) ? (ub[(size_t)nbr * 512 + c] - cv) : cv;
  X[(size_t)col * 1024 + ch] = (bf16)v;
}

// ds_sp post: Y [512][512], max over kk=2 -> d_out sp region [b][512][32]
__global__ void __launch_bounds__(256) apply_dssp_max(
    const float* __restrict__ Y, const float* __restrict__ stat,
    const float* __restrict__ g, const float* __restrict__ bt, float* __restrict__ out)
{
  int idx = blockIdx.x * 256 + threadIdx.x;
  if (idx >= 8 * 512 * 32) return;
  int c = idx & 511, t = idx >> 9;
  int b = t >> 5, m = t & 31;
  float mn = stat[c], r = stat[512 + c], gg = g[c], bb = bt[c];
  float v0 = gelu_f((Y[(size_t)c * 512 + t * 2 + 0] - mn) * r * gg + bb);
  float v1 = gelu_f((Y[(size_t)c * 512 + t * 2 + 1] - mn) * r * gg + bb);
  out[((size_t)b * 512 + c) * 32 + m] = fmaxf(v0, v1);
}

// ds_dn edge features with the reference's exact (sp c)-split flat-index math
__global__ void __launch_bounds__(256) build_edge_ds_dn(
    const float* __restrict__ dnf, const int* __restrict__ fidx,
    const int* __restrict__ knnf, bf16* __restrict__ X)
{
  long long idx = (long long)blockIdx.x * 256 + threadIdx.x;
  if (idx >= 16384LL * 512) return;
  int ch = (int)(idx & 511); int col = (int)(idx >> 9);
  int kk = col & 1, pp = (col >> 1) & 31, m = (col >> 6) & 31, b = col >> 11;
  int t = b * 32 + m;
  int nbr = knnf[t * 2 + kk], ctr = fidx[t];
  int flat = pp * 512 + ch;
  float v;
  if (flat < 8192) {
    int p = flat >> 8, c = flat & 255;
    v = dnf[(((size_t)b * 64 + nbr) * 32 + p) * 256 + c]
      - dnf[(((size_t)b * 64 + ctr) * 32 + p) * 256 + c];
  } else {
    int f2 = flat - 8192;
    int p = f2 >> 8, c = f2 & 255;
    v = dnf[(((size_t)b * 64 + ctr) * 32 + p) * 256 + c];
  }
  X[(size_t)col * 512 + ch] = (bf16)v;
}

// ds_dn post: Y [256][16384], max over kk=2 -> dnpre [b][m][pp][256]
__global__ void __launch_bounds__(256) apply_dsdn_max(
    const float* __restrict__ Y, const float* __restrict__ stat,
    const float* __restrict__ g, const float* __restrict__ bt, float* __restrict__ dnpre)
{
  long long idx = (long long)blockIdx.x * 256 + threadIdx.x;
  if (idx >= 8LL * 32 * 32 * 256) return;
  int c = (int)(idx & 255);
  int t = (int)(idx >> 8);   // (b*32+m)*32+pp
  float mn = stat[c], r = stat[512 + c], gg = g[c], bb = bt[c];
  float v0 = gelu_f((Y[(size_t)c * 16384 + (size_t)t * 2 + 0] - mn) * r * gg + bb);
  float v1 = gelu_f((Y[(size_t)c * 16384 + (size_t)t * 2 + 1] - mn) * r * gg + bb);
  dnpre[(size_t)t * 256 + c] = fmaxf(v0, v1);
}

// im2col for (1,3) stride-2 conv: Xds[(b*32+m)*16+q][i*3+kk]
__global__ void __launch_bounds__(256) im2col_ds(
    const float* __restrict__ dnpre, bf16* __restrict__ X)
{
  long long idx = (long long)blockIdx.x * 256 + threadIdx.x;
  if (idx >= 4096LL * 768) return;
  int k = (int)(idx % 768); int col = (int)(idx / 768);
  int kk = k % 3, i = k / 3;
  int q = col & 15, m = (col >> 4) & 31, b = col >> 9;
  int pp = 2 * q + kk - 1;
  float v = (pp >= 0 && pp < 32)
      ? dnpre[(((size_t)b * 32 + m) * 32 + pp) * 256 + i] : 0.f;
  X[idx] = (bf16)v;
}

// final: Y [256][4096] -> BN+GELU -> d_out dn region [b][256][32][16]
__global__ void __launch_bounds__(256) apply_final(
    const float* __restrict__ Y, const float* __restrict__ stat,
    const float* __restrict__ g, const float* __restrict__ bt, float* __restrict__ out)
{
  long long idx = (long long)blockIdx.x * 256 + threadIdx.x;
  if (idx >= 256LL * 4096) return;
  int c = (int)(idx & 255); int col = (int)(idx >> 8);
  int q = col & 15, m = (col >> 4) & 31, b = col >> 9;
  float v = Y[(size_t)c * 4096 + col];
  v = gelu_f((v - stat[c]) * stat[512 + c] * g[c] + bt[c]);
  out[(((size_t)b * 256 + c) * 32 + m) * 16 + q] = v;
}

// ---------------------------------------------------------------------------
extern "C" void kernel_launch(void* const* d_in, const int* in_sizes, int n_in,
                              void* d_out, int out_size, void* d_ws, size_t ws_size,
                              hipStream_t stream)
{
  (void)in_sizes; (void)n_in; (void)out_size; (void)ws_size;

  const float* sparse_fea = (const float*)d_in[0];
  const float* dense_fea  = (const float*)d_in[1];
  const float* stk_coor   = (const float*)d_in[2];
  const float* d2s_tw  = (const float*)d_in[3];
  const float* d2s_tg  = (const float*)d_in[5];
  const float* d2s_tbt = (const float*)d_in[6];
  const float* d2s_ew  = (const float*)d_in[7];
  const float* d2s_eg  = (const float*)d_in[9];
  const float* d2s_ebt = (const float*)d_in[10];
  const float* s2d_w   = (const float*)d_in[11];
  const float* s2d_g   = (const float*)d_in[13];
  const float* s2d_bt  = (const float*)d_in[14];
  const float* spu_w   = (const float*)d_in[15];
  const float* spu_g   = (const float*)d_in[17];
  const float* spu_bt  = (const float*)d_in[18];
  const float* dnu_w   = (const float*)d_in[19];
  const float* dnu_g   = (const float*)d_in[21];
  const float* dnu_bt  = (const float*)d_in[22];
  const float* ds_spw  = (const float*)d_in[23];
  const float* ds_spg  = (const float*)d_in[25];
  const float* ds_spbt = (const float*)d_in[26];
  const float* ds_dnw  = (const float*)d_in[27];
  const float* ds_dng  = (const float*)d_in[29];
  const float* ds_dnbt = (const float*)d_in[30];
  const float* ds_dsw  = (const float*)d_in[31];
  const float* ds_dsg  = (const float*)d_in[33];
  const float* ds_dsbt = (const float*)d_in[34];

  float* out_sp   = (float*)d_out;                     // [8][512][32]
  float* out_dn   = (float*)d_out + 131072;            // [8][256][32][16]
  float* out_coor = (float*)d_out + 131072 + 1048576;  // [8][32][32]

  char* p = (char*)d_ws;
  auto alloc = [&](size_t bytes) -> void* {
    void* r = (void*)p; p += (bytes + 255) & ~(size_t)255; return r;
  };

  bf16*  W_T    = (bf16*)alloc(128 * 384 * sizeof(bf16));
  bf16*  W_E    = (bf16*)alloc(256 * 384 * sizeof(bf16));
  bf16*  W_S2D  = (bf16*)alloc(128 * 384 * sizeof(bf16));
  bf16*  W_SPU  = (bf16*)alloc(512 * 512 * sizeof(bf16));
  bf16*  W_DNU  = (bf16*)alloc(256 * 256 * sizeof(bf16));
  bf16*  W_DSSP = (bf16*)alloc(512 * 1024 * sizeof(bf16));
  bf16*  W_DSDN = (bf16*)alloc(256 * 512 * sizeof(bf16));
  bf16*  W_DSDS = (bf16*)alloc(256 * 768 * sizeof(bf16));
  bf16*  XT   = (bf16*)alloc(16384ULL * 384 * sizeof(bf16));
  float* YT   = (float*)alloc(128ULL * 16384 * sizeof(float));
  bf16*  XC   = (bf16*)alloc(512ULL * 384 * sizeof(bf16));
  float* YC   = (float*)alloc(256ULL * 512 * sizeof(float));
  float* USP  = (float*)alloc(8ULL * 64 * 256 * sizeof(float));
  bf16*  XD   = (bf16*)alloc(16384ULL * 384 * sizeof(bf16));
  float* YD   = (float*)alloc(128ULL * 16384 * sizeof(float));
  bf16*  UDN  = (bf16*)alloc(8ULL * 2048 * 128 * sizeof(bf16));
  float* NN   = (float*)alloc(8ULL * 2048 * sizeof(float));
  float* GBUF = (float*)alloc(2048ULL * 2048 * sizeof(float));
  int*   IDXD = (int*)alloc(8ULL * 2048 * 10 * sizeof(int));
  bf16*  XE   = (bf16*)alloc(20480ULL * 256 * sizeof(bf16));
  bf16*  YE   = (bf16*)alloc(256ULL * 163840 * sizeof(bf16));
  float* DNF  = (float*)alloc(8ULL * 64 * 32 * 256 * sizeof(float));
  int*   IDXS = (int*)alloc(8ULL * 64 * 2 * sizeof(int));
  bf16*  XSP  = (bf16*)alloc(1024ULL * 512 * sizeof(bf16));
  float* YSP  = (float*)alloc(512ULL * 1024 * sizeof(float));
  float* USP2 = (float*)alloc(8ULL * 64 * 512 * sizeof(float));
  int*   FIDX = (int*)alloc(8ULL * 32 * sizeof(int));
  int*   KNNF = (int*)alloc(8ULL * 32 * 2 * sizeof(int));
  bf16*  XSP2 = (bf16*)alloc(512ULL * 1024 * sizeof(bf16));
  float* YSP2 = (float*)alloc(512ULL * 512 * sizeof(float));
  bf16*  XDN2 = (bf16*)alloc(16384ULL * 512 * sizeof(bf16));
  float* YDN2 = (float*)alloc(256ULL * 16384 * sizeof(float));
  float* DNPRE= (float*)alloc(8ULL * 32 * 32 * 256 * sizeof(float));
  bf16*  XDS  = (bf16*)alloc(4096ULL * 768 * sizeof(bf16));
  float* YDS  = (float*)alloc(256ULL * 4096 * sizeof(float));
  float* STAT = (float*)alloc(1024 * sizeof(float));

  auto g1 = [](long long n) -> dim3 { return dim3((unsigned)((n + 255) / 256)); };
  dim3 B(256);

  auto gemm = [&](const bf16* A, const bf16* Bm, void* C,
                  int M, int N, int K, int ldc, int cbf) {
    int tiles = (M >> 4) * (N >> 6);
    gemm_bf16<<<dim3((unsigned)((tiles + 7) / 8)), B, 0, stream>>>(A, Bm, C, M, N, K, ldc, cbf);
  };

  // ---- pack weights to bf16 (bias terms cancel under BatchNorm) ----
  cvt_bf16<<<g1(128 * 384), B, 0, stream>>>(d2s_tw, W_T, 128 * 384);
  cvt_bf16<<<g1(256 * 384), B, 0, stream>>>(d2s_ew, W_E, 256 * 384);
  cvt_bf16<<<g1(128 * 384), B, 0, stream>>>(s2d_w, W_S2D, 128 * 384);
  cvt_bf16<<<g1(512 * 512), B, 0, stream>>>(spu_w, W_SPU, 512 * 512);
  cvt_bf16<<<g1(256 * 256), B, 0, stream>>>(dnu_w, W_DNU, 256 * 256);
  cvt_bf16<<<g1(512 * 1024), B, 0, stream>>>(ds_spw, W_DSSP, 512 * 1024);
  cvt_bf16<<<g1(256 * 512), B, 0, stream>>>(ds_dnw, W_DSDN, 256 * 512);
  cvt_bf16<<<g1(256 * 768), B, 0, stream>>>(ds_dsw, W_DSDS, 256 * 768);

  // ---- DenseToSparse: temporal conv -> BN+GELU -> max over points ----
  im2col_t<<<g1(16384LL * 384), B, 0, stream>>>(dense_fea, XT);
  gemm(W_T, XT, YT, 128, 16384, 384, 16384, 0);
  bn_stats_f32<<<dim3(128), B, 0, stream>>>(YT, 16384, STAT);
  apply_t_max<<<g1(512 * 128), B, 0, stream>>>(YT, STAT, d2s_tg, d2s_tbt, XC);
  pack_sparse_xc<<<g1(512 * 256), B, 0, stream>>>(sparse_fea, XC);
  gemm(W_E, XC, YC, 256, 512, 384, 512, 0);
  bn_stats_f32<<<dim3(256), B, 0, stream>>>(YC, 512, STAT);
  apply_bn_gelu_T_f32<<<g1(256LL * 512), B, 0, stream>>>(YC, 256, 512, STAT, d2s_eg, d2s_ebt, USP);

  // ---- SparseToDense ----
  build_xd<<<g1(16384LL * 384), B, 0, stream>>>(dense_fea, sparse_fea, XD);
  gemm(W_S2D, XD, YD, 128, 16384, 384, 16384, 0);
  bn_stats_f32<<<dim3(128), B, 0, stream>>>(YD, 16384, STAT);
  apply_bn_gelu_T_bf16<<<g1(128LL * 16384), B, 0, stream>>>(YD, 128, 16384, STAT, s2d_g, s2d_bt, UDN);

  // ---- Dense EdgeConv: per-batch gram GEMM + top-10, then edge GEMM ----
  sqnorm_k<<<g1(16384), B, 0, stream>>>(UDN, NN);
  for (int b = 0; b < 8; b++) {
    const bf16* udn_b = UDN + (size_t)b * 2048 * 128;
    gemm(udn_b, udn_b, GBUF, 2048, 2048, 128, 2048, 0);
    topk10<<<dim3(8), B, 0, stream>>>(GBUF, NN + (size_t)b * 2048, IDXD + (size_t)b * 20480);
  }
  for (int b = 0; b < 8; b++) {
    const bf16* udn_b = UDN + (size_t)b * 2048 * 128;
    build_edge_dn<<<g1(20480LL * 256), B, 0, stream>>>(udn_b, IDXD + (size_t)b * 20480, XE);
    gemm(W_DNU, XE, YE + (size_t)b * 20480, 256, 20480, 256, 163840, 1);
  }
  bn_stats_bf16<<<dim3(256), B, 0, stream>>>(YE, 163840, STAT);
  apply_dn_max<<<g1(8LL * 2048 * 256), B, 0, stream>>>(YE, STAT, dnu_g, dnu_bt, DNF);

  // ---- Sparse EdgeConv ----
  knn_sp<<<g1(512), B, 0, stream>>>(USP, IDXS);
  build_edge_sp_k<<<g1(1024LL * 512), B, 0, stream>>>(USP, IDXS, XSP);
  gemm(W_SPU, XSP, YSP, 512, 1024, 512, 1024, 0);
  bn_stats_f32<<<dim3(512), B, 0, stream>>>(YSP, 1024, STAT);
  apply_sp_max<<<g1(8 * 64 * 512), B, 0, stream>>>(YSP, STAT, spu_g, spu_bt, USP2);

  // ---- DownSample: FPS + coord kNN + gathers ----
  fps_k<<<dim3(8), dim3(64), 0, stream>>>(stk_coor, FIDX);
  knn_coor_k<<<g1(256), B, 0, stream>>>(stk_coor, FIDX, KNNF);
  gather_coor_k<<<g1(8192), B, 0, stream>>>(stk_coor, FIDX, out_coor);

  // sparse branch of downsample -> sp_out
  build_edge_ds_sp<<<g1(512LL * 1024), B, 0, stream>>>(USP2, FIDX, KNNF, XSP2);
  gemm(W_DSSP, XSP2, YSP2, 512, 512, 1024, 512, 0);
  bn_stats_f32<<<dim3(512), B, 0, stream>>>(YSP2, 512, STAT);
  apply_dssp_max<<<g1(8 * 512 * 32), B, 0, stream>>>(YSP2, STAT, ds_spg, ds_spbt, out_sp);

  // dense branch of downsample
  build_edge_ds_dn<<<g1(16384LL * 512), B, 0, stream>>>(DNF, FIDX, KNNF, XDN2);
  gemm(W_DSDN, XDN2, YDN2, 256, 16384, 512, 16384, 0);
  bn_stats_f32<<<dim3(256), B, 0, stream>>>(YDN2, 16384, STAT);
  apply_dsdn_max<<<g1(8LL * 32 * 32 * 256), B, 0, stream>>>(YDN2, STAT, ds_dng, ds_dnbt, DNPRE);

  // stride-2 (1,3) conv + BN + GELU -> dn output
  im2col_ds<<<g1(4096LL * 768), B, 0, stream>>>(DNPRE, XDS);
  gemm(W_DSDS, XDS, YDS, 256, 4096, 768, 4096, 0);
  bn_stats_f32<<<dim3(256), B, 0, stream>>>(YDS, 4096, STAT);
  apply_final<<<g1(256LL * 4096), B, 0, stream>>>(YDS, STAT, ds_dsg, ds_dsbt, out_dn);
}
